// CSPNet_78864189489978
// MI455X (gfx1250) — compile-verified, matrix operand order checked by reading, not run on hardware
//
#include <hip/hip_runtime.h>
#include <hip/hip_bf16.h>
#include <stdint.h>

typedef __attribute__((ext_vector_type(16))) _Float16 v16h;
typedef __attribute__((ext_vector_type(8)))  float    v8f;
typedef __attribute__((ext_vector_type(4)))  int      v4i;
using half_t = _Float16;

#define NGRAPH 128
#define NATOM  32
#define HID    128
#define LAT    256
#define NLAYER 4
#define NODES  (NGRAPH*NATOM)      // 4096
#define EDGES  (NODES*NATOM)       // 131072
#define KDEMB  64                  // 60 padded to 64
#define KIN    (HID+LAT)           // 384

// ---------------- workspace layout (bytes) ----------------
constexpr size_t alignB(size_t x){ return (x + 255) & ~size_t(255); }
constexpr size_t OFF_LATWT = 0;                      constexpr size_t SZ_LATWT = (size_t)HID*KIN*2;
constexpr size_t OFF_W1AT  = alignB(OFF_LATWT+SZ_LATWT);  constexpr size_t SZ_W1AT = (size_t)NLAYER*HID*HID*2;
constexpr size_t OFF_W1BT  = alignB(OFF_W1AT+SZ_W1AT);    constexpr size_t SZ_W1BT = SZ_W1AT;
constexpr size_t OFF_W1DT  = alignB(OFF_W1BT+SZ_W1BT);    constexpr size_t SZ_W1DT = (size_t)NLAYER*HID*KDEMB*2;
constexpr size_t OFF_W2T   = alignB(OFF_W1DT+SZ_W1DT);    constexpr size_t SZ_W2T  = (size_t)NLAYER*HID*HID*2;
constexpr size_t OFF_NW1T  = alignB(OFF_W2T+SZ_W2T);      constexpr size_t SZ_NW1T = (size_t)NLAYER*HID*256*2;
constexpr size_t OFF_NW2T  = alignB(OFF_NW1T+SZ_NW1T);    constexpr size_t SZ_NW2T = (size_t)NLAYER*HID*HID*2;
constexpr size_t OFF_H     = alignB(OFF_NW2T+SZ_NW2T);    constexpr size_t SZ_H    = (size_t)NODES*HID*4;
constexpr size_t OFF_HH    = alignB(OFF_H+SZ_H);          constexpr size_t SZ_HH   = (size_t)NODES*HID*2;
constexpr size_t OFF_DEMB  = alignB(OFF_HH+SZ_HH);        constexpr size_t SZ_DEMB = (size_t)EDGES*KDEMB*2;
constexpr size_t OFF_LATIP = alignB(OFF_DEMB+SZ_DEMB);    constexpr size_t SZ_LATIP= (size_t)NGRAPH*9*4;
constexpr size_t OFF_HA    = alignB(OFF_LATIP+SZ_LATIP);  constexpr size_t SZ_HA   = (size_t)NGRAPH*NATOM*HID*4;
constexpr size_t OFF_HB    = alignB(OFF_HA+SZ_HA);        constexpr size_t SZ_HB   = SZ_HA;
constexpr size_t OFF_AGG   = alignB(OFF_HB+SZ_HB);        constexpr size_t SZ_AGG  = (size_t)NODES*HID*4;
constexpr size_t OFF_GFEAT = alignB(OFF_AGG+SZ_AGG);      constexpr size_t SZ_GFEAT= (size_t)NGRAPH*HID*4;

// ---------------- helpers ----------------
__device__ __forceinline__ float silu_f(float x){ return x / (1.f + __expf(-x)); }

__device__ __forceinline__ v8f wmma_f16(v16h a, v16h b, v8f c){
  return __builtin_amdgcn_wmma_f32_16x16x32_f16(false, a, false, b, (short)0, c, false, false);
}

// Async bulk copy 16B global -> LDS (ASYNCcnt-tracked, no VGPR round trip).
// lds_off = LDS byte offset (low 32 bits of flat shared pointer).
__device__ __forceinline__ void async_copy16(uint32_t lds_off, const void* g){
  asm volatile("global_load_async_to_lds_b128 %0, %1, off"
               :: "v"(lds_off), "v"(g) : "memory");
}
__device__ __forceinline__ void wait_async0(){
  asm volatile("s_wait_asynccnt 0x0" ::: "memory");
}

// Load one 16x32 f16 A/B fragment from a row-major tile (A: rows=M, B: rows=N of W^T).
// ISA layout: lane L holds row L%16; K elems 0..7 at kb, 8..15 at kb+16, kb = (L>=16)?8:0.
__device__ __forceinline__ v16h load_frag(const half_t* tile, int stride){
  const int lane = threadIdx.x & 31;
  const int m    = lane & 15;
  const int kb   = (lane & 16) ? 8 : 0;
  const half_t* p = tile + m*stride + kb;
  union U { v16h v; v4i q[2]; } f;
  f.q[0] = *(const v4i*)(p);
  f.q[1] = *(const v4i*)(p + 16);
  return f.v;
}

// ---------------- weight conversion (f32 -> f16 transposed) ----------------
__global__ void k_convert(const float* latent_w, const float* edge_w1, const float* edge_w2,
                          const float* node_w1, const float* node_w2,
                          half_t* latwT, half_t* w1aT, half_t* w1bT, half_t* w1dT,
                          half_t* w2T, half_t* nw1T, half_t* nw2T){
  const int tid = blockIdx.x*blockDim.x + threadIdx.x;
  const int nth = gridDim.x*blockDim.x;
  for (int i = tid; i < HID*KIN; i += nth){ int n=i/KIN, k=i%KIN; latwT[i] = (half_t)latent_w[k*HID+n]; }
  for (int l = 0; l < NLAYER; l++){
    const float* w1 = edge_w1 + (size_t)l*325*HID;
    half_t* a = w1aT + (size_t)l*HID*HID;
    half_t* b = w1bT + (size_t)l*HID*HID;
    half_t* d = w1dT + (size_t)l*HID*KDEMB;
    half_t* w2= w2T  + (size_t)l*HID*HID;
    half_t* n1= nw1T + (size_t)l*HID*256;
    half_t* n2= nw2T + (size_t)l*HID*HID;
    for (int i = tid; i < HID*HID; i += nth){ int n=i/HID,k=i%HID; a[i]=(half_t)w1[k*HID+n]; b[i]=(half_t)w1[(128+k)*HID+n]; }
    for (int i = tid; i < HID*KDEMB; i += nth){ int n=i/KDEMB,k=i%KDEMB; d[i] = (k<60)?(half_t)w1[(265+k)*HID+n]:(half_t)0.f; }
    for (int i = tid; i < HID*HID; i += nth){ int n=i/HID,k=i%HID; w2[i]=(half_t)edge_w2[(size_t)l*HID*HID + k*HID+n]; }
    for (int i = tid; i < HID*256; i += nth){ int n=i/256,k=i%256; n1[i]=(half_t)node_w1[(size_t)l*256*HID + k*HID+n]; }
    for (int i = tid; i < HID*HID; i += nth){ int n=i/HID,k=i%HID; n2[i]=(half_t)node_w2[(size_t)l*HID*HID + k*HID+n]; }
  }
}

// ---------------- lattice inner products ----------------
__global__ void k_lat(const float* lattices, float* lat_ips){
  int idx = blockIdx.x*blockDim.x + threadIdx.x;
  if (idx >= NGRAPH*9) return;
  int b = idx/9, i = (idx%9)/3, k = idx%3;
  const float* L = lattices + b*9;
  lat_ips[idx] = L[i*3+0]*L[k*3+0] + L[i*3+1]*L[k*3+1] + L[i*3+2]*L[k*3+2];
}

// ---------------- sinusoidal distance embedding (f16, padded to 64) ----------------
__global__ void k_demb(const float* frac, half_t* demb){
  int e = blockIdx.x*blockDim.x + threadIdx.x;
  if (e >= EDGES) return;
  int g = e >> 10, el = e & 1023, i = el >> 5, j = el & 31;
  int ns = g*NATOM + i, nd = g*NATOM + j;
  half_t* out = demb + (size_t)e*KDEMB;
  float d[3];
  #pragma unroll
  for (int c = 0; c < 3; c++){ float v = frac[nd*3+c] - frac[ns*3+c]; v -= floorf(v); d[c] = v; }
  const float TWO_PI = 6.283185307179586f;
  #pragma unroll
  for (int c = 0; c < 3; c++)
    for (int f = 0; f < 10; f++){
      float x = d[c]*TWO_PI*(float)f;
      out[c*10+f]      = (half_t)__sinf(x);
      out[30 + c*10+f] = (half_t)__cosf(x);
    }
  out[60]=out[61]=out[62]=out[63]=(half_t)0.f;
}

// ---------------- h0 = concat(emb, t) @ latent_w + b ----------------
__global__ void __launch_bounds__(128) k_node_init(const float* emb, const float* tlat,
                          const float* latent_b, const int* atom_types,
                          const half_t* latwT, float* h, half_t* hH){
  __shared__ half_t As[16*(KIN+8)];         // stride 392
  const int tid = threadIdx.x, wv = tid>>5, lane = tid&31;
  const int row0 = blockIdx.x*16;
  for (int i = tid; i < 16*KIN; i += blockDim.x){
    int r = i/KIN, c = i%KIN, node = row0 + r;
    float v = (c < HID) ? emb[(atom_types[node]-1)*HID + c] : tlat[(node>>5)*LAT + (c-HID)];
    As[r*(KIN+8)+c] = (half_t)v;
  }
  __syncthreads();
  const int cm = (lane>>4)<<3, cn = lane&15;
  for (int nt = wv*2; nt < wv*2+2; nt++){
    v8f c; float bv = latent_b[nt*16+cn];
    #pragma unroll
    for (int r = 0; r < 8; r++) c[r] = bv;
    for (int ks = 0; ks < KIN/32; ks++)
      c = wmma_f16(load_frag(As + ks*32, KIN+8),
                   load_frag(latwT + (size_t)(nt*16)*KIN + ks*32, KIN), c);
    #pragma unroll
    for (int r = 0; r < 8; r++){
      int node = row0+cm+r, col = nt*16+cn;
      h[(size_t)node*HID+col] = c[r]; hH[(size_t)node*HID+col] = (half_t)c[r];
    }
  }
}

// ---------------- per-graph HA = H@W1a, HB' = H@W1b + lat@W1c + b1 ----------------
__global__ void __launch_bounds__(128) k_pre(const half_t* hH, const half_t* w1aT, const half_t* w1bT,
                     const float* w1c, const float* b1, const float* lat_ips,
                     float* HA, float* HB){
  __shared__ float latc[HID];
  const int g = blockIdx.x, tid = threadIdx.x, wv = tid>>5, lane = tid&31;
  {
    const float* lp = lat_ips + g*9;
    float s = b1[tid];
    #pragma unroll
    for (int p = 0; p < 9; p++) s += lp[p]*w1c[p*HID + tid];
    latc[tid] = s;
  }
  __syncthreads();
  const half_t* hg = hH + (size_t)g*NATOM*HID;
  const int cm = (lane>>4)<<3, cn = lane&15;
  for (int job = wv; job < 32; job += 4){
    const int mat = job & 1, rt = (job>>1)&1, nt = job>>2;
    v8f c = {};
    if (mat){ float v = latc[nt*16+cn];
      #pragma unroll
      for (int r = 0; r < 8; r++) c[r] = v; }
    const half_t* wT = mat ? w1bT : w1aT;
    for (int ks = 0; ks < 4; ks++)
      c = wmma_f16(load_frag(hg + rt*16*HID + ks*32, HID),
                   load_frag(wT + (size_t)(nt*16)*HID + ks*32, HID), c);
    float* out = (mat ? HB : HA) + (size_t)g*NATOM*HID;
    #pragma unroll
    for (int r = 0; r < 8; r++) out[(rt*16+cm+r)*HID + nt*16+cn] = c[r];
  }
}

// ---------------- fused edge MLP + scatter-mean ----------------
__global__ void __launch_bounds__(128) k_edge(const half_t* demb, const half_t* w1dT, const half_t* w2T,
                      const float* HA, const float* HB, const float* b2, float* agg){
  __shared__ half_t sW1d[HID*KDEMB];       // 16 KB
  __shared__ half_t sW2 [HID*HID];         // 32 KB
  __shared__ float  sHA [8*HID];           //  4 KB
  __shared__ float  sHB [NATOM*HID];       // 16 KB
  __shared__ float  sAgg[8*HID];           //  4 KB
  __shared__ half_t sE1 [4][16*136];       // 17 KB (per-wave e1 staging)
  const int g = blockIdx.x >> 2, i0 = (blockIdx.x & 3)*8;
  const int tid = threadIdx.x, wv = tid>>5, lane = tid&31;

  // ---- async bulk staging: global -> LDS, ASYNCcnt-tracked ----
  {
    const uint32_t l_w1d = (uint32_t)(uintptr_t)&sW1d[0];
    const uint32_t l_w2  = (uint32_t)(uintptr_t)&sW2[0];
    const uint32_t l_ha  = (uint32_t)(uintptr_t)&sHA[0];
    const uint32_t l_hb  = (uint32_t)(uintptr_t)&sHB[0];
    const char* gw1d = (const char*)w1dT;
    const char* gw2  = (const char*)w2T;
    const char* gha  = (const char*)(HA + ((size_t)g*NATOM + i0)*HID);
    const char* ghb  = (const char*)(HB + (size_t)g*NATOM*HID);
    for (int i = tid; i < HID*KDEMB*2/16; i += blockDim.x)  // 1024 x 16B
      async_copy16(l_w1d + i*16, gw1d + (size_t)i*16);
    for (int i = tid; i < HID*HID*2/16; i += blockDim.x)    // 2048 x 16B
      async_copy16(l_w2 + i*16, gw2 + (size_t)i*16);
    for (int i = tid; i < 8*HID*4/16; i += blockDim.x)      //  256 x 16B
      async_copy16(l_ha + i*16, gha + (size_t)i*16);
    for (int i = tid; i < NATOM*HID*4/16; i += blockDim.x)  // 1024 x 16B
      async_copy16(l_hb + i*16, ghb + (size_t)i*16);
  }
  for (int i = tid; i < 8*HID; i += blockDim.x) sAgg[i] = 0.f;
  wait_async0();
  __syncthreads();

  const int cm = (lane>>4)<<3, cn = lane&15;
  float bv8[8];
  #pragma unroll
  for (int nt = 0; nt < 8; nt++) bv8[nt] = b2[nt*16+cn];

  for (int t = wv*4; t < wv*4+4; t++){
    const int il = t>>1, jh = t&1, j0 = jh*16;
    const size_t erow = (size_t)g*1024 + (size_t)(i0+il)*32 + j0;
    const half_t* dtile = demb + erow*KDEMB;
    __builtin_prefetch(dtile, 0, 3);
    v16h a0 = load_frag(dtile, KDEMB);
    v16h a1 = load_frag(dtile + 32, KDEMB);
    half_t* stg = sE1[wv];
    for (int nt = 0; nt < 8; nt++){
      v8f c;
      float hav = sHA[il*HID + nt*16+cn];
      #pragma unroll
      for (int r = 0; r < 8; r++) c[r] = hav + sHB[(j0+cm+r)*HID + nt*16+cn];
      c = wmma_f16(a0, load_frag(sW1d + nt*16*KDEMB,      KDEMB), c);
      c = wmma_f16(a1, load_frag(sW1d + nt*16*KDEMB + 32, KDEMB), c);
      #pragma unroll
      for (int r = 0; r < 8; r++) stg[(cm+r)*136 + nt*16+cn] = (half_t)silu_f(c[r]);
    }
    __asm__ volatile("s_wait_dscnt 0" ::: "memory");   // RAW on per-wave LDS staging
    v16h e0 = load_frag(stg,      136);
    v16h e1 = load_frag(stg+32,   136);
    v16h e2 = load_frag(stg+64,   136);
    v16h e3 = load_frag(stg+96,   136);
    for (int nt = 0; nt < 8; nt++){
      v8f c;
      #pragma unroll
      for (int r = 0; r < 8; r++) c[r] = bv8[nt];
      c = wmma_f16(e0, load_frag(sW2 + nt*16*HID,      HID), c);
      c = wmma_f16(e1, load_frag(sW2 + nt*16*HID + 32, HID), c);
      c = wmma_f16(e2, load_frag(sW2 + nt*16*HID + 64, HID), c);
      c = wmma_f16(e3, load_frag(sW2 + nt*16*HID + 96, HID), c);
      float part = 0.f;
      #pragma unroll
      for (int r = 0; r < 8; r++) part += silu_f(c[r]);
      atomicAdd(&sAgg[il*HID + nt*16+cn], part);       // ds_add_f32
    }
  }
  __syncthreads();
  float* aout = agg + ((size_t)g*NATOM + i0)*HID;
  for (int i = tid; i < 8*HID; i += blockDim.x) aout[i] = sAgg[i]*(1.f/32.f);
}

// ---------------- node MLP + residual ----------------
__global__ void __launch_bounds__(128) k_node(float* h, half_t* hH, const float* agg,
                      const half_t* nw1T, const half_t* nw2T,
                      const float* nb1, const float* nb2){
  __shared__ half_t A1[16*264];
  __shared__ half_t M1[16*136];
  const int row0 = blockIdx.x*16, tid = threadIdx.x, wv = tid>>5, lane = tid&31;
  for (int i = tid; i < 16*256; i += blockDim.x){
    int r = i>>8, c = i&255;
    float v = (c < HID) ? h[(size_t)(row0+r)*HID + c] : agg[(size_t)(row0+r)*HID + (c-HID)];
    A1[r*264+c] = (half_t)v;
  }
  __syncthreads();
  const int cm = (lane>>4)<<3, cn = lane&15;
  for (int nt = wv*2; nt < wv*2+2; nt++){
    v8f c; float bv = nb1[nt*16+cn];
    #pragma unroll
    for (int r = 0; r < 8; r++) c[r] = bv;
    for (int ks = 0; ks < 8; ks++)
      c = wmma_f16(load_frag(A1 + ks*32, 264),
                   load_frag(nw1T + (size_t)(nt*16)*256 + ks*32, 256), c);
    #pragma unroll
    for (int r = 0; r < 8; r++) M1[(cm+r)*136 + nt*16+cn] = (half_t)silu_f(c[r]);
  }
  __syncthreads();
  for (int nt = wv*2; nt < wv*2+2; nt++){
    v8f c; float bv = nb2[nt*16+cn];
    #pragma unroll
    for (int r = 0; r < 8; r++) c[r] = bv;
    for (int ks = 0; ks < 4; ks++)
      c = wmma_f16(load_frag(M1 + ks*32, 136),
                   load_frag(nw2T + (size_t)(nt*16)*HID + ks*32, HID), c);
    #pragma unroll
    for (int r = 0; r < 8; r++){
      int node = row0+cm+r, col = nt*16+cn;
      float v = h[(size_t)node*HID+col] + silu_f(c[r]);
      h[(size_t)node*HID+col] = v; hH[(size_t)node*HID+col] = (half_t)v;
    }
  }
}

// ---------------- output heads ----------------
__global__ void k_head1(const float* h, const float* coord_w, float* out, float* gfeat){
  int idx = blockIdx.x*blockDim.x + threadIdx.x;
  if (idx < NODES*3){
    int node = idx/3, c = idx%3;
    float s = 0.f;
    for (int k = 0; k < HID; k++) s += h[(size_t)node*HID+k]*coord_w[k*3+c];
    out[NGRAPH*9 + idx] = s;
  } else if (idx < NODES*3 + NGRAPH*HID){
    int r = idx - NODES*3, g = r/HID, f = r%HID;
    float s = 0.f;
    for (int n = 0; n < NATOM; n++) s += h[(size_t)(g*NATOM+n)*HID + f];
    gfeat[g*HID+f] = s*(1.f/32.f);
  }
}
__global__ void k_head2(const float* gfeat, const float* lattice_w, const float* lattices, float* out){
  int idx = blockIdx.x*blockDim.x + threadIdx.x;
  if (idx >= NGRAPH*9) return;
  int g = idx/9, i = (idx%9)/3, k = idx%3;
  float s = 0.f;
  for (int j = 0; j < 3; j++){
    float t = 0.f;
    for (int q = 0; q < HID; q++) t += gfeat[g*HID+q]*lattice_w[q*9 + i*3 + j];
    s += t * lattices[g*9 + j*3 + k];
  }
  out[idx] = s;
}

// ---------------- launch ----------------
extern "C" void kernel_launch(void* const* d_in, const int* in_sizes, int n_in,
                              void* d_out, int out_size, void* d_ws, size_t ws_size,
                              hipStream_t stream){
  const float* t_lat     = (const float*)d_in[0];
  const float* frac      = (const float*)d_in[1];
  const float* lattices  = (const float*)d_in[2];
  const float* emb       = (const float*)d_in[3];
  const float* latent_w  = (const float*)d_in[4];
  const float* latent_b  = (const float*)d_in[5];
  const float* edge_w1   = (const float*)d_in[6];
  const float* edge_b1   = (const float*)d_in[7];
  const float* edge_w2   = (const float*)d_in[8];
  const float* edge_b2   = (const float*)d_in[9];
  const float* node_w1   = (const float*)d_in[10];
  const float* node_b1   = (const float*)d_in[11];
  const float* node_w2   = (const float*)d_in[12];
  const float* node_b2   = (const float*)d_in[13];
  const float* coord_w   = (const float*)d_in[14];
  const float* lattice_w = (const float*)d_in[15];
  const int*   atom_types= (const int*)d_in[16];
  float* out = (float*)d_out;
  char*  ws  = (char*)d_ws;

  half_t* latwT = (half_t*)(ws + OFF_LATWT);
  half_t* w1aT  = (half_t*)(ws + OFF_W1AT);
  half_t* w1bT  = (half_t*)(ws + OFF_W1BT);
  half_t* w1dT  = (half_t*)(ws + OFF_W1DT);
  half_t* w2T   = (half_t*)(ws + OFF_W2T);
  half_t* nw1T  = (half_t*)(ws + OFF_NW1T);
  half_t* nw2T  = (half_t*)(ws + OFF_NW2T);
  float*  h     = (float*) (ws + OFF_H);
  half_t* hH    = (half_t*)(ws + OFF_HH);
  half_t* demb  = (half_t*)(ws + OFF_DEMB);
  float*  latip = (float*) (ws + OFF_LATIP);
  float*  HA    = (float*) (ws + OFF_HA);
  float*  HB    = (float*) (ws + OFF_HB);
  float*  agg   = (float*) (ws + OFF_AGG);
  float*  gfeat = (float*) (ws + OFF_GFEAT);

  k_convert<<<256, 256, 0, stream>>>(latent_w, edge_w1, edge_w2, node_w1, node_w2,
                                     latwT, w1aT, w1bT, w1dT, w2T, nw1T, nw2T);
  k_lat<<<(NGRAPH*9 + 255)/256, 256, 0, stream>>>(lattices, latip);
  k_demb<<<EDGES/256, 256, 0, stream>>>(frac, demb);
  k_node_init<<<NODES/16, 128, 0, stream>>>(emb, t_lat, latent_b, atom_types, latwT, h, hH);

  for (int l = 0; l < NLAYER; l++){
    k_pre<<<NGRAPH, 128, 0, stream>>>(hH,
        w1aT + (size_t)l*HID*HID, w1bT + (size_t)l*HID*HID,
        edge_w1 + (size_t)l*325*HID + 256*HID, edge_b1 + l*HID, latip, HA, HB);
    k_edge<<<NGRAPH*4, 128, 0, stream>>>(demb,
        w1dT + (size_t)l*HID*KDEMB, w2T + (size_t)l*HID*HID,
        HA, HB, edge_b2 + l*HID, agg);
    k_node<<<NODES/16, 128, 0, stream>>>(h, hH, agg,
        nw1T + (size_t)l*HID*256, nw2T + (size_t)l*HID*HID,
        node_b1 + l*HID, node_b2 + l*HID);
  }

  k_head1<<<(NODES*3 + NGRAPH*HID + 255)/256, 256, 0, stream>>>(h, coord_w, out, gfeat);
  k_head2<<<(NGRAPH*9 + 255)/256, 256, 0, stream>>>(gfeat, lattice_w, lattices, out);
}
